// Model_32057635897637
// MI455X (gfx1250) — matrix operand order repaired and checked by reading, not compiled, not run on hardware
//
#include <hip/hip_runtime.h>

#define BB    4096
#define LL    50
#define BLR   (BB * LL)      // 204800 rows
#define DD    256
#define DBERT 768
#define HH    4
#define DHD   64
#define TM    128            // block tile rows
#define TN    64             // block tile cols
#define TK    64             // K per LDS stage

typedef unsigned short u16t;
typedef unsigned int   u32t;
typedef __attribute__((ext_vector_type(16))) __bf16 v16bf;
typedef __attribute__((ext_vector_type(8)))  __bf16 v8bf;
typedef __attribute__((ext_vector_type(8)))  float  v8f;
typedef __attribute__((ext_vector_type(4)))  u32t   v4u;
typedef __attribute__((ext_vector_type(8)))  u32t   v8u;

__device__ __forceinline__ u16t f2bf(float f) {
  __bf16 h = (__bf16)f;              // native v_cvt_*bf16_f32 (RNE)
  u16t r;
  __builtin_memcpy(&r, &h, 2);
  return r;
}
__device__ __forceinline__ float bf2f(u16t h) {
  return __uint_as_float(((u32t)h) << 16);
}

__device__ __forceinline__ v8f wmma_bf16(v16bf a, v16bf b, v8f c) {
  return __builtin_amdgcn_wmma_f32_16x16x32_bf16(false, a, false, b, (short)0, c,
                                                 false, false);
}

// A fragment: 16x32 bf16 (ISA 7.12.2): lane<16 -> M=lane, elems0-7=K[ks+half*8..],
// elems8-15=K[ks+16+half*8..]
__device__ __forceinline__ v16bf frag_a(const u16t* t, int lda, int mi, int ks, int lane) {
  int m    = mi * 16 + (lane & 15);
  int half = lane >> 4;
  const __bf16* r0 = (const __bf16*)(t + m * lda + ks + half * 8);
  const __bf16* r1 = (const __bf16*)(t + m * lda + ks + 16 + half * 8);
  v16bf a;
#pragma unroll
  for (int i = 0; i < 8; ++i) a[i] = r0[i];
#pragma unroll
  for (int i = 0; i < 8; ++i) a[i + 8] = r1[i];
  return a;
}
// B fragment: 32x16 bf16, lane -> col N=nt*16+lane%16, elems0-7=K[ks+half*8..],
// elems8-15=K[ks+16+half*8..]
__device__ __forceinline__ v16bf frag_b(const u16t* t, int ldb, int nt, int ks, int lane) {
  const __bf16* r = (const __bf16*)(t + nt * 16 + (lane & 15));
  int half = lane >> 4;
  v16bf b;
#pragma unroll
  for (int i = 0; i < 16; ++i) b[i] = r[(ks + (i < 8 ? half * 8 + i : 16 + half * 8 + (i - 8))) * ldb];
  return b;
}

// ---- CDNA5 async global->LDS copy (ASYNCcnt path, CDNA5 ISA) ----
__device__ __forceinline__ void async_b128(u32t lds_byte, const void* gaddr) {
  asm volatile("global_load_async_to_lds_b128 %0, %1, off"
               :: "v"(lds_byte), "v"(gaddr) : "memory");
}
__device__ __forceinline__ void wait_async() {
  asm volatile("s_wait_asynccnt 0x0" ::: "memory");
}

// ---- CDNA5 Tensor Data Mover: 2-D bf16 tile -> LDS (TENSORcnt path, §7/§8) ----
// dim0 = contiguous elements per row, dim1 = rows, stride in elements.
__device__ __forceinline__ void tdm_load_2d(u32t lds_byte, const void* gaddr,
                                            int dim0, int dim1, int stride) {
  unsigned long long ga = (unsigned long long)gaddr;
  v4u g0;
  g0[0] = 1u;                                            // count=1, user D#
  g0[1] = lds_byte;                                      // lds_addr (bytes)
  g0[2] = (u32t)ga;                                      // global_addr[31:0]
  g0[3] = (u32t)((ga >> 32) & 0x1FFFFFFu) | (2u << 30);  // addr[56:32] | type=2
  v8u g1;
  g1[0] = (1u << 16);                                    // data_size=1 -> 2 bytes
  g1[1] = ((u32t)dim0 & 0xFFFFu) << 16;                  // tensor_dim0[15:0]
  g1[2] = (((u32t)dim0 >> 16) & 0xFFFFu) |
          (((u32t)dim1 & 0xFFFFu) << 16);                // dim0 hi | tensor_dim1 lo
  g1[3] = (((u32t)dim1 >> 16) & 0xFFFFu) |
          (((u32t)dim0 & 0xFFFFu) << 16);                // dim1 hi | tile_dim0
  g1[4] = (u32t)dim1 & 0xFFFFu;                          // tile_dim1 | tile_dim2=0
  g1[5] = (u32t)stride;                                  // tensor_dim0_stride[31:0]
  g1[6] = 0u;
  g1[7] = 0u;
  asm volatile("tensor_load_to_lds %0, %1" :: "s"(g0), "s"(g1) : "memory");
}
__device__ __forceinline__ void wait_tensor() {
  __builtin_amdgcn_s_wait_tensorcnt(0);
}

// stage TM x TK bf16 tile (async): thread t -> row t>>1, 32-elem half (t&1)
__device__ __forceinline__ void stage_a_async(u32t ldsA, const u16t* A, long grow0,
                                              int ld, int kt, int tid) {
  int r = tid >> 1;
  int c = (tid & 1) * 32;
  const u16t* src = A + (grow0 + r) * (long)ld + kt + c;
  u32t dst = ldsA + (u32t)(r * TK + c) * 2;
#pragma unroll
  for (int i = 0; i < 4; ++i) async_b128(dst + i * 16, src + i * 8);
}

// one K-stage of WMMAs: wave 'mi' computes 16 x 64 with 8 WMMAs (A-frag reuse x4)
__device__ __forceinline__ void mma_step(const u16t* As, const u16t* Bs,
                                         int mi, int lane, v8f c[4]) {
#pragma unroll
  for (int ks = 0; ks < TK; ks += 32) {
    v16bf a = frag_a(As, TK, mi, ks, lane);
#pragma unroll
    for (int nt = 0; nt < 4; ++nt)
      c[nt] = wmma_bf16(a, frag_b(Bs, TN, nt, ks, lane), c[nt]);
  }
}

// generic bf16 GEMM mainloop: A [rows x K] row-major bf16, W [K x N] bf16
// A tile staged via ASYNC per-lane copies; W tile staged via one TDM descriptor.
__device__ __forceinline__ void gemm_main_bf16(const u16t* A, int K, const u16t* W,
                                               int N, long row0, int n0, v8f c[4]) {
  extern __shared__ char smem[];
  const u16t* As = (const u16t*)smem;
  const u16t* Bs = (const u16t*)(smem + 16384);
  u32t base = __builtin_amdgcn_groupstaticsize();
  int tid = threadIdx.x;
  int lane = tid & 31, mi = tid >> 5;
  for (int kt = 0; kt < K; kt += TK) {
    stage_a_async(base, A, row0, K, kt, tid);
    if (tid < 32) {                         // wave 0 drives the TDM
      tdm_load_2d(base + 16384, W + (long)kt * N + n0, TN, TK, N);
      wait_tensor();
    }
    wait_async();
    __syncthreads();
    mma_step(As, Bs, mi, lane, c);
    __syncthreads();
  }
}

// gather-GEMM mainloop: A rows gathered from f32 table (convert to bf16 in regs)
__device__ __forceinline__ void gemm_main_gather(const float* table, long arow,
                                                 const u16t* W, int n0, v8f c[4]) {
  extern __shared__ char smem[];
  u16t* As = (u16t*)smem;
  const u16t* Bs = (const u16t*)(smem + 16384);
  u32t base = __builtin_amdgcn_groupstaticsize();
  int tid = threadIdx.x;
  int lane = tid & 31, mi = tid >> 5;
  int r = tid >> 1, cc = (tid & 1) * 32;
  u16t* dstA = As + r * TK + cc;
  for (int kt = 0; kt < DBERT; kt += TK) {
    if (tid < 32) {
      tdm_load_2d(base + 16384, W + (long)kt * DD + n0, TN, TK, DD);
      wait_tensor();
    }
    const float* s = table + arow + kt + cc;
#pragma unroll
    for (int i = 0; i < 4; ++i) {
      float4 f0 = *(const float4*)(s + i * 8);
      float4 f1 = *(const float4*)(s + i * 8 + 4);
      v8bf h;
      h[0] = (__bf16)f0.x; h[1] = (__bf16)f0.y; h[2] = (__bf16)f0.z; h[3] = (__bf16)f0.w;
      h[4] = (__bf16)f1.x; h[5] = (__bf16)f1.y; h[6] = (__bf16)f1.z; h[7] = (__bf16)f1.w;
      *(v8bf*)(dstA + i * 8) = h;
    }
    __syncthreads();
    mma_step(As, Bs, mi, lane, c);
    __syncthreads();
  }
}

// ---------------- f32 -> bf16 weight conversion ----------------
__global__ void __launch_bounds__(256) k_cvt(const float* __restrict__ src,
                                             u16t* __restrict__ dst, int n) {
  int i = blockIdx.x * 256 + threadIdx.x;
  if (i < n) dst[i] = f2bf(src[i]);
}

// ---------------- hist: embed[nids] @ Wproj + b + pe -> bf16 ----------------
__global__ void __launch_bounds__(256) k_gather_hist(
    const float* __restrict__ table, const int* __restrict__ nids,
    const u16t* __restrict__ Wb, const float* __restrict__ bias,
    const float* __restrict__ pe, u16t* __restrict__ outb) {
  long row0 = (long)blockIdx.x * TM;
  int  n0   = blockIdx.y * TN;
  int  tid  = threadIdx.x;
  long arow = (long)nids[row0 + (tid >> 1)] * DBERT;
  v8f c[4] = {};
  gemm_main_gather(table, arow, Wb, n0, c);
  int lane = tid & 31, mi = tid >> 5;
  int half = lane >> 4, nl = lane & 15;
#pragma unroll
  for (int r = 0; r < 8; ++r) {
    int row = (int)row0 + mi * 16 + r + 8 * half;
    const float* perow = pe + (row % LL) * DD;
#pragma unroll
    for (int nt = 0; nt < 4; ++nt) {
      int col = n0 + nt * 16 + nl;
      outb[(long)row * DD + col] = f2bf(c[nt][r] + bias[col] + perow[col]);
    }
  }
}

// ---------------- news: embed[impr] @ Wproj + b -> f32 ----------------
__global__ void __launch_bounds__(256) k_gather_news(
    const float* __restrict__ table, const int* __restrict__ nids,
    const u16t* __restrict__ Wb, const float* __restrict__ bias,
    float* __restrict__ outf) {
  long row0 = (long)blockIdx.x * TM;
  int  n0   = blockIdx.y * TN;
  int  tid  = threadIdx.x;
  long arow = (long)nids[row0 + (tid >> 1)] * DBERT;
  v8f c[4] = {};
  gemm_main_gather(table, arow, Wb, n0, c);
  int lane = tid & 31, mi = tid >> 5;
  int half = lane >> 4, nl = lane & 15;
#pragma unroll
  for (int r = 0; r < 8; ++r) {
    int row = (int)row0 + mi * 16 + r + 8 * half;
#pragma unroll
    for (int nt = 0; nt < 4; ++nt) {
      int col = n0 + nt * 16 + nl;
      outf[(long)row * DD + col] = c[nt][r] + bias[col];
    }
  }
}

// ---------------- QKV projections (WMMA), scatter to [B,H,...] ----------------
__global__ void __launch_bounds__(256) k_qkv(
    const u16t* __restrict__ x,
    const u16t* __restrict__ wq, const u16t* __restrict__ wk, const u16t* __restrict__ wv,
    const float* __restrict__ bq, const float* __restrict__ bk, const float* __restrict__ bv,
    u16t* __restrict__ qo, u16t* __restrict__ kto, u16t* __restrict__ vo) {
  int which = blockIdx.z;
  const u16t*  W    = which == 0 ? wq : (which == 1 ? wk : wv);
  const float* bias = which == 0 ? bq : (which == 1 ? bk : bv);
  long row0 = (long)blockIdx.x * TM;
  int  n0   = blockIdx.y * TN;
  v8f c[4] = {};
  gemm_main_bf16(x, DD, W, DD, row0, n0, c);
  int tid = threadIdx.x;
  int lane = tid & 31, mi = tid >> 5;
  int half = lane >> 4, nl = lane & 15;
#pragma unroll
  for (int r = 0; r < 8; ++r) {
    int row = (int)row0 + mi * 16 + r + 8 * half;
    int b = row / LL, l = row % LL;
#pragma unroll
    for (int nt = 0; nt < 4; ++nt) {
      int col = n0 + nt * 16 + nl;
      int h = col >> 6, dh = col & 63;
      u16t hv = f2bf(c[nt][r] + bias[col]);
      long bh = (long)b * HH + h;
      if (which == 1)      kto[(bh * DHD + dh) * LL + l] = hv;
      else if (which == 0) qo[(bh * LL + l) * DHD + dh]  = hv;
      else                 vo[(bh * LL + l) * DHD + dh]  = hv;
    }
  }
}

// ---------------- fused attention per (b,h): QK^T -> softmax -> @V -------------
__global__ void __launch_bounds__(256) k_attn(
    const u16t* __restrict__ qb, const u16t* __restrict__ ktb, const u16t* __restrict__ vb,
    const int* __restrict__ hlen, u16t* __restrict__ ob) {
  extern __shared__ char smem[];
  u16t*  Qs = (u16t*)smem;             // 64x64
  u16t*  Ks = (u16t*)(smem + 8192);    // 64x64 (k^T: rows=dh, cols=key)
  u16t*  Vs = (u16t*)(smem + 16384);   // 64x64
  float* Ss = (float*)(smem + 24576);  // 64x64
  u32t base = __builtin_amdgcn_groupstaticsize();
  int bh = blockIdx.x;
  int b = bh >> 2, h = bh & 3;
  int tid = threadIdx.x;
  int len = hlen[b];
  const u16t* qsrc = qb  + (long)bh * LL * DHD;
  const u16t* vsrc = vb  + (long)bh * LL * DHD;
  const u16t* ksrc = ktb + (long)bh * DHD * LL;
  {
    int r = tid >> 2, c = (tid & 3) * 16;
    u32t doff = (u32t)(r * 64 + c) * 2;
    if (r < LL) {
      async_b128(base + doff,              qsrc + r * DHD + c);
      async_b128(base + doff + 16,         qsrc + r * DHD + c + 8);
      async_b128(base + 16384 + doff,      vsrc + r * DHD + c);
      async_b128(base + 16384 + doff + 16, vsrc + r * DHD + c + 8);
    } else {
      uint4 z = make_uint4(0u, 0u, 0u, 0u);
      *(uint4*)(Qs + r * 64 + c) = z; *(uint4*)(Qs + r * 64 + c + 8) = z;
      *(uint4*)(Vs + r * 64 + c) = z; *(uint4*)(Vs + r * 64 + c + 8) = z;
    }
#pragma unroll
    for (int j = 0; j < 16; ++j) {
      int col = c + j;
      Ks[r * 64 + col] = (col < LL) ? ksrc[r * LL + col] : (u16t)0;
    }
  }
  wait_async();
  __syncthreads();
  int lane = tid & 31, wave = tid >> 5, mi = wave & 3, nt0 = (wave >> 2) * 2;
  v8f c0 = {}; v8f c1 = {};
#pragma unroll
  for (int s = 0; s < 2; ++s) {
    v16bf a = frag_a(Qs, 64, mi, s * 32, lane);
    c0 = wmma_bf16(a, frag_b(Ks, 64, nt0, s * 32, lane), c0);
    c1 = wmma_bf16(a, frag_b(Ks, 64, nt0 + 1, s * 32, lane), c1);
  }
  int half = lane >> 4, nl = lane & 15;
#pragma unroll
  for (int r = 0; r < 8; ++r) {
    int m = mi * 16 + r + 8 * half;
    Ss[m * 64 + nt0 * 16 + nl]       = c0[r] * 0.5f;   // ATT_SCALE = 1/sqrt(H)
    Ss[m * 64 + (nt0 + 1) * 16 + nl] = c1[r] * 0.5f;
  }
  __syncthreads();
  if (tid < 64) {                      // masked softmax per query row -> bf16 in Qs
    float mx = -1e30f;
    for (int j = 0; j < len; ++j) mx = fmaxf(mx, Ss[tid * 64 + j]);
    float sum = 0.f;
    for (int j = 0; j < len; ++j) sum += expf(Ss[tid * 64 + j] - mx);
    float inv = 1.f / sum;
    for (int j = 0; j < 64; ++j) {
      float w = (j < len) ? expf(Ss[tid * 64 + j] - mx) * inv : 0.f;
      Qs[tid * 64 + j] = f2bf(w);
    }
  }
  __syncthreads();
  v8f d0 = {}; v8f d1 = {};
#pragma unroll
  for (int s = 0; s < 2; ++s) {
    v16bf a = frag_a(Qs, 64, mi, s * 32, lane);
    d0 = wmma_bf16(a, frag_b(Vs, 64, nt0, s * 32, lane), d0);
    d1 = wmma_bf16(a, frag_b(Vs, 64, nt0 + 1, s * 32, lane), d1);
  }
#pragma unroll
  for (int r = 0; r < 8; ++r) {
    int l = mi * 16 + r + 8 * half;
    if (l < LL) {
      long bas = ((long)b * LL + l) * DD + h * DHD;
      ob[bas + nt0 * 16 + nl]       = f2bf(d0[r]);
      ob[bas + (nt0 + 1) * 16 + nl] = f2bf(d1[r]);
    }
  }
}

// ------------- pooling scores: tanh(o@Wa1+ba1)@Wa2 -> partial[row][nblk] -------
__global__ void __launch_bounds__(256) k_pool(
    const u16t* __restrict__ ob, const u16t* __restrict__ wa1,
    const float* __restrict__ ba1, const float* __restrict__ wa2,
    float* __restrict__ partial) {
  extern __shared__ char smem[];
  float* S2 = (float*)(smem + 24576);  // TM x TN
  long row0 = (long)blockIdx.x * TM;
  int  n0   = blockIdx.y * TN;
  v8f c[4] = {};
  gemm_main_bf16(ob, DD, wa1, DD, row0, n0, c);
  int tid = threadIdx.x;
  int lane = tid & 31, mi = tid >> 5;
  int half = lane >> 4, nl = lane & 15;
#pragma unroll
  for (int r = 0; r < 8; ++r) {
    int m = mi * 16 + r + 8 * half;
#pragma unroll
    for (int nt = 0; nt < 4; ++nt) {
      int col = n0 + nt * 16 + nl;
      S2[m * TN + nt * 16 + nl] = tanhf(c[nt][r] + ba1[col]) * wa2[col];
    }
  }
  __syncthreads();
  if (tid < TM) {                      // deterministic serial row reduction
    float s = 0.f;
    for (int j = 0; j < TN; ++j) s += S2[tid * TN + j];
    partial[(row0 + tid) * 4 + blockIdx.y] = s;
  }
}

// ------------- per-batch softmax pooling: user = sum_l w_l * o[b,l,:] -----------
__global__ void __launch_bounds__(256) k_user(
    const float* __restrict__ partial, const int* __restrict__ hlen,
    const u16t* __restrict__ ob, float* __restrict__ user) {
  __shared__ float sl[LL];
  __shared__ float wl[LL];
  __shared__ float ms[2];
  int b = blockIdx.x, tid = threadIdx.x;
  int len = hlen[b];
  if (tid < LL) {
    const float* p = partial + ((long)b * LL + tid) * 4;
    sl[tid] = p[0] + p[1] + p[2] + p[3];
  }
  __syncthreads();
  if (tid == 0) {
    float mx = -1e30f;
    for (int l = 0; l < len; ++l) mx = fmaxf(mx, sl[l]);
    float sum = 0.f;
    for (int l = 0; l < len; ++l) sum += expf(sl[l] - mx);
    ms[0] = mx; ms[1] = sum;
  }
  __syncthreads();
  if (tid < LL) wl[tid] = (tid < len) ? expf(sl[tid] - ms[0]) / ms[1] : 0.f;
  __syncthreads();
  float acc = 0.f;
  for (int l = 0; l < len; ++l)
    acc += wl[l] * bf2f(ob[((long)b * LL + l) * DD + tid]);
  user[(long)b * DD + tid] = acc;
}

// ------------- comb = [u, n, u+n, u*n] -> bf16 ------------------
__global__ void __launch_bounds__(256) k_comb(const float* __restrict__ user,
                                              const float* __restrict__ news,
                                              u16t* __restrict__ comb) {
  int i = blockIdx.x * 256 + threadIdx.x;
  int b = i >> 8, d = i & 255;
  float u = user[i], n = news[i];
  long base = (long)b * 1024 + d;
  comb[base]       = f2bf(u);
  comb[base + 256] = f2bf(n);
  comb[base + 512] = f2bf(u + n);
  comb[base + 768] = f2bf(u * n);
}

// ------------- generic FC layer (WMMA) + leaky ReLU ------------------
__global__ void __launch_bounds__(256) k_fc(
    const u16t* __restrict__ A, const u16t* __restrict__ W,
    const float* __restrict__ bias, u16t* __restrict__ out,
    int K, int N, int leaky) {
  long row0 = (long)blockIdx.x * TM;
  int  n0   = blockIdx.y * TN;
  v8f c[4] = {};
  gemm_main_bf16(A, K, W, N, row0, n0, c);
  int tid = threadIdx.x;
  int lane = tid & 31, mi = tid >> 5;
  int half = lane >> 4, nl = lane & 15;
#pragma unroll
  for (int r = 0; r < 8; ++r) {
    long row = row0 + mi * 16 + r + 8 * half;
#pragma unroll
    for (int nt = 0; nt < 4; ++nt) {
      int col = n0 + nt * 16 + nl;
      float v = c[nt][r] + bias[col];
      if (leaky) v = v > 0.f ? v : 0.01f * v;
      out[row * N + col] = f2bf(v);
    }
  }
}

// ------------- final 128->2 head (VALU) ------------------
__global__ void __launch_bounds__(256) k_out(const u16t* __restrict__ h2,
                                             const float* __restrict__ Wf3,
                                             const float* __restrict__ bf3,
                                             float* __restrict__ out) {
  int b = blockIdx.x * 256 + threadIdx.x;
  if (b >= BB) return;
  float a0 = bf3[0], a1 = bf3[1];
  for (int i = 0; i < 128; ++i) {
    float hv = bf2f(h2[(long)b * 128 + i]);
    a0 += hv * Wf3[i * 2];
    a1 += hv * Wf3[i * 2 + 1];
  }
  out[b * 2] = a0;
  out[b * 2 + 1] = a1;
}

// ------------- NLL loss, single-block deterministic reduce ---------
__global__ void __launch_bounds__(256) k_loss(const float* __restrict__ out,
                                              const int* __restrict__ label,
                                              float* __restrict__ lossp) {
  __shared__ float red[256];
  int tid = threadIdx.x;
  float acc = 0.f;
  for (int b = tid; b < BB; b += 256) {
    float o0 = out[2 * b], o1 = out[2 * b + 1];
    float m = fmaxf(o0, o1);
    float lse = m + logf(expf(o0 - m) + expf(o1 - m));
    float lp = (label[b] ? o1 : o0) - lse;
    acc -= lp;
  }
  red[tid] = acc;
  __syncthreads();
  for (int s = 128; s > 0; s >>= 1) {
    if (tid < s) red[tid] += red[tid + s];
    __syncthreads();
  }
  if (tid == 0) lossp[0] = red[0] / (float)BB;
}

extern "C" void kernel_launch(void* const* d_in, const int* in_sizes, int n_in,
                              void* d_out, int out_size, void* d_ws, size_t ws_size,
                              hipStream_t stream) {
  (void)in_sizes; (void)n_in; (void)out_size; (void)ws_size;
  const float* table = (const float*)d_in[0];
  const float* Wproj = (const float*)d_in[1];
  const float* bproj = (const float*)d_in[2];
  const float* Wq    = (const float*)d_in[3];
  const float* bq    = (const float*)d_in[4];
  const float* Wk    = (const float*)d_in[5];
  const float* bk    = (const float*)d_in[6];
  const float* Wv    = (const float*)d_in[7];
  const float* bv    = (const float*)d_in[8];
  const float* Wa1   = (const float*)d_in[9];
  const float* ba1   = (const float*)d_in[10];
  const float* Wa2   = (const float*)d_in[11];
  // d_in[12] = ba2: softmax shift-invariant, unused
  const float* Wf1   = (const float*)d_in[13];
  const float* bf1   = (const float*)d_in[14];
  const float* Wf2   = (const float*)d_in[15];
  const float* bf2   = (const float*)d_in[16];
  const float* Wf3   = (const float*)d_in[17];
  const float* bf3   = (const float*)d_in[18];
  const float* pe    = (const float*)d_in[19];
  const int* hist_nids = (const int*)d_in[20];
  const int* hist_len  = (const int*)d_in[21];
  const int* impr      = (const int*)d_in[22];
  const int* label     = (const int*)d_in[23];
  float* out = (float*)d_out;

  char* ws = (char*)d_ws;
  size_t off = 0;
  auto carve = [&](size_t bytes) -> char* {
    char* p = ws + off;
    off = (off + bytes + 255) & ~(size_t)255;
    return p;
  };
  u16t* wProjB = (u16t*)carve((size_t)DBERT * DD * 2);
  u16t* wQB    = (u16t*)carve((size_t)DD * DD * 2);
  u16t* wKB    = (u16t*)carve((size_t)DD * DD * 2);
  u16t* wVB    = (u16t*)carve((size_t)DD * DD * 2);
  u16t* wA1B   = (u16t*)carve((size_t)DD * DD * 2);
  u16t* wF1B   = (u16t*)carve((size_t)1024 * 512 * 2);
  u16t* wF2B   = (u16t*)carve((size_t)512 * 128 * 2);
  u16t* xo     = (u16t*)carve((size_t)BLR * DD * 2);      // x, then reused as o
  u16t* qB     = (u16t*)carve((size_t)BB * HH * LL * DHD * 2);
  u16t* ktB    = (u16t*)carve((size_t)BB * HH * DHD * LL * 2);
  u16t* vB     = (u16t*)carve((size_t)BB * HH * LL * DHD * 2);
  float* part  = (float*)carve((size_t)BLR * 4 * 4);
  float* userF = (float*)carve((size_t)BB * DD * 4);
  float* newsF = (float*)carve((size_t)BB * DD * 4);
  u16t* combB  = (u16t*)carve((size_t)BB * 1024 * 2);
  u16t* h1B    = (u16t*)carve((size_t)BB * 512 * 2);
  u16t* h2B    = (u16t*)carve((size_t)BB * 128 * 2);

  // 1) weight conversion f32 -> bf16
  k_cvt<<<dim3((DBERT * DD + 255) / 256), 256, 0, stream>>>(Wproj, wProjB, DBERT * DD);
  k_cvt<<<dim3((DD * DD + 255) / 256), 256, 0, stream>>>(Wq, wQB, DD * DD);
  k_cvt<<<dim3((DD * DD + 255) / 256), 256, 0, stream>>>(Wk, wKB, DD * DD);
  k_cvt<<<dim3((DD * DD + 255) / 256), 256, 0, stream>>>(Wv, wVB, DD * DD);
  k_cvt<<<dim3((DD * DD + 255) / 256), 256, 0, stream>>>(Wa1, wA1B, DD * DD);
  k_cvt<<<dim3((1024 * 512 + 255) / 256), 256, 0, stream>>>(Wf1, wF1B, 1024 * 512);
  k_cvt<<<dim3((512 * 128 + 255) / 256), 256, 0, stream>>>(Wf2, wF2B, 512 * 128);

  // 2) x = embed[history] @ Wproj + b + pe (bf16); news = embed[impr] @ Wproj + b (f32)
  k_gather_hist<<<dim3(BLR / TM, DD / TN), 256, 24576, stream>>>(
      table, hist_nids, wProjB, bproj, pe, xo);
  k_gather_news<<<dim3(BB / TM, DD / TN), 256, 24576, stream>>>(
      table, impr, wProjB, bproj, newsF);
  // 3) QKV
  k_qkv<<<dim3(BLR / TM, DD / TN, 3), 256, 24576, stream>>>(
      xo, wQB, wKB, wVB, bq, bk, bv, qB, ktB, vB);
  // 4) fused attention -> o (overwrites x buffer)
  k_attn<<<dim3(BB * HH), 256, 40960, stream>>>(qB, ktB, vB, hist_len, xo);
  // 5) pooling scores
  k_pool<<<dim3(BLR / TM, DD / TN), 256, 24576 + TM * TN * 4, stream>>>(
      xo, wA1B, ba1, Wa2, part);
  // 6) softmax-pooled user vector
  k_user<<<dim3(BB), 256, 0, stream>>>(part, hist_len, xo, userF);
  // 7) comb features
  k_comb<<<dim3(BB), 256, 0, stream>>>(userF, newsF, combB);
  // 8) FC1 (1024->512, leaky), FC2 (512->128, leaky)
  k_fc<<<dim3(BB / TM, 512 / TN), 256, 24576, stream>>>(combB, wF1B, bf1, h1B, 1024, 512, 1);
  k_fc<<<dim3(BB / TM, 128 / TN), 256, 24576, stream>>>(h1B, wF2B, bf2, h2B, 512, 128, 1);
  // 9) head + loss
  k_out<<<dim3((BB + 255) / 256), 256, 0, stream>>>(h2B, Wf3, bf3, out);
  k_loss<<<dim3(1), 256, 0, stream>>>(out, label, out + (size_t)BB * 2);
}
